// FeatureRefineModule_16192026706268
// MI455X (gfx1250) — compile-verified, matrix operand order checked by reading, not run on hardware
//
#include <hip/hip_runtime.h>

typedef __bf16 bf16_t;
typedef bf16_t v16bf __attribute__((ext_vector_type(16)));
typedef float  v8f   __attribute__((ext_vector_type(8)));

#define CC  256
#define HH  192
#define WW  192
#define HWW (HH*WW)

#define SBS 40   // B-tile ci-stride (bf16): 80 B rows -> 16B-aligned, bank-conflict-free

struct FragBF { union { unsigned int u[8]; v16bf v; }; };

static __device__ __forceinline__ unsigned lds_addr32(const void* p) {
  // generic shared pointer: high 32 bits = LDS aperture, low 32 bits = LDS offset
  return (unsigned)(unsigned long long)p;
}

static __device__ __forceinline__ void async_cp_b128(unsigned lds_dst, const void* gsrc) {
  asm volatile("global_load_async_to_lds_b128 %0, %1, off"
               :: "v"(lds_dst), "v"(gsrc) : "memory");
}

static __device__ __forceinline__ void wait_async0() {
  asm volatile("s_wait_asynccnt 0x0" ::: "memory");
}

// ---------------- prep: fp32 -> bf16 conversions ----------------
__global__ __launch_bounds__(256) void k_cvt_x(const float* __restrict__ x,
                                               bf16_t* __restrict__ xb, int n) {
  int i = blockIdx.x * 256 + threadIdx.x;
  if (i < n) xb[i] = (bf16_t)x[i];
}

// src (co*256+ci)*5 + k  ->  dst (co*5+k)*256 + ci   (contiguous ci per tap)
__global__ __launch_bounds__(256) void k_cvt_w5(const float* __restrict__ w,
                                                bf16_t* __restrict__ wb) {
  int i = blockIdx.x * 256 + threadIdx.x;
  if (i < CC * CC * 5) {
    int k = i % 5; int rem = i / 5;
    int ci = rem & 255; int co = rem >> 8;
    wb[(co * 5 + k) * CC + ci] = (bf16_t)w[i];
  }
}

__global__ __launch_bounds__(256) void k_cvt_w1(const float* __restrict__ w,
                                                bf16_t* __restrict__ wb) {
  int i = blockIdx.x * 256 + threadIdx.x;
  if (i < CC * CC) wb[i] = (bf16_t)w[i];
}

// ---------------- conv 1x5 (pad w=2): t1 = conv(x, w15) + b15 ----------------
// Tile: 64 co x 64 w x 4 h-rows. Weight chunk amortized over 4 rows.
// 8 waves: wv&3 = co-sub (16 rows), wv>>2 = n-sub (32 px => 2 WMMA tiles).
__global__ __launch_bounds__(256)
void k_conv1x5(const bf16_t* __restrict__ xb, const bf16_t* __restrict__ w15b,
               const float* __restrict__ b15, bf16_t* __restrict__ t1b) {
  __shared__ __align__(16) bf16_t sA[5 * 64 * 32];      // [kw][co64][ci32]  20480 B
  __shared__ __align__(16) bf16_t sB[4 * 68 * SBS];     // [row][px][ci32]   21760 B

  const int w0  = blockIdx.x * 64;
  const int h0  = blockIdx.y * 4;
  const int b   = blockIdx.z >> 2;
  const int coB = blockIdx.z & 3;
  const int tid = threadIdx.x;
  const int wv   = tid >> 5;
  const int lane = tid & 31;
  const int half = lane >> 4;
  const int lx   = lane & 15;
  const int mSub = wv & 3, nSub = wv >> 2;
  const int coRow0 = coB * 64 + mSub * 16;

  v8f acc[4][2];
  for (int row = 0; row < 4; ++row)
    for (int t = 0; t < 2; ++t)
      for (int r = 0; r < 8; ++r)
        acc[row][t][r] = b15[coRow0 + half * 8 + r];

  const unsigned sAbase = lds_addr32(&sA[0]);

  for (int kk = 0; kk < 8; ++kk) {
    __syncthreads();
    // weights: async global->LDS, 1280 x b128 (layout preserving)
    for (int j = tid; j < 1280; j += 256) {
      int cip4 = j & 3;
      int rem = j >> 2;                    // 0..319
      int co = rem / 5, kw = rem - co * 5;
      const bf16_t* src =
          w15b + ((size_t)((coB * 64 + co) * 5 + kw) * CC + kk * 32 + cip4 * 8);
      unsigned dst = sAbase + (unsigned)(((kw * 64 + co) * 32 + cip4 * 8) * 2);
      async_cp_b128(dst, src);
    }
    // input strips: 4 rows x 68 px x 32 ci, transposed to [px][ci], zero-pad in w
    for (int idx = tid; idx < 4 * 68 * 32; idx += 256) {
      int ci  = idx & 31;
      int rem = idx >> 5;                  // 0..271
      int px  = rem % 68;
      int row = rem / 68;
      int w = w0 - 2 + px;
      bf16_t v = (bf16_t)0.0f;
      if ((unsigned)w < (unsigned)WW)
        v = xb[(size_t)(b * CC + kk * 32 + ci) * HWW + (h0 + row) * WW + w];
      sB[(row * 68 + px) * SBS + ci] = v;
    }
    wait_async0();
    __syncthreads();

    for (int kw = 0; kw < 5; ++kw) {
      FragBF a;
      const unsigned int* ap =
          (const unsigned int*)(sA + (kw * 64 + mSub * 16 + lx) * 32);
      #pragma unroll
      for (int i = 0; i < 8; ++i) {
        int kb = (i >> 2) * 16 + half * 8 + (i & 3) * 2;
        a.u[i] = ap[kb >> 1];
      }
      #pragma unroll
      for (int row = 0; row < 4; ++row) {
        #pragma unroll
        for (int t = 0; t < 2; ++t) {
          FragBF bb;
          int px = (nSub * 2 + t) * 16 + lx + kw;          // tap = window shift
          const unsigned int* bp =
              (const unsigned int*)(sB + (row * 68 + px) * SBS);
          #pragma unroll
          for (int i = 0; i < 8; ++i) {
            int kb = (i >> 2) * 16 + half * 8 + (i & 3) * 2;
            bb.u[i] = bp[kb >> 1];
          }
          acc[row][t] = __builtin_amdgcn_wmma_f32_16x16x32_bf16(
              false, a.v, false, bb.v, (short)0, acc[row][t], false, false);
        }
      }
    }
  }

  #pragma unroll
  for (int row = 0; row < 4; ++row)
    #pragma unroll
    for (int t = 0; t < 2; ++t) {
      int w = w0 + (nSub * 2 + t) * 16 + lx;
      #pragma unroll
      for (int r = 0; r < 8; ++r) {
        int co = coRow0 + half * 8 + r;
        t1b[(size_t)(b * CC + co) * HWW + (h0 + row) * WW + w] =
            (bf16_t)acc[row][t][r];
      }
    }
}

// ------- conv 5x1 (pad h=2) over t1  +  fused conv 1x1 over x  -> feat -------
// Tile: 64 co x 64 w x 2 h-rows; the 2 rows share 4 of 6 staged input rows.
__global__ __launch_bounds__(256)
void k_conv5x1_1x1(const bf16_t* __restrict__ t1b, const bf16_t* __restrict__ xb,
                   const bf16_t* __restrict__ w51b, const bf16_t* __restrict__ w11b,
                   const float* __restrict__ b51, const float* __restrict__ b11,
                   float* __restrict__ feat) {
  __shared__ __align__(16) bf16_t sA51[5 * 64 * 32];    // [kh][co64][ci32]  20480 B
  __shared__ __align__(16) bf16_t sA11[64 * 32];        // [co64][ci32]       4096 B
  __shared__ __align__(16) bf16_t sB51[6 * 64 * SBS];   // [row6][n64][ci32] 30720 B
  __shared__ __align__(16) bf16_t sB11[2 * 64 * 36];    // [row2][n64][ci32]  9216 B
  // total 64512 B

  const int w0  = blockIdx.x * 64;
  const int h0  = blockIdx.y * 2;
  const int b   = blockIdx.z >> 2;
  const int coB = blockIdx.z & 3;
  const int tid = threadIdx.x;
  const int wv   = tid >> 5;
  const int lane = tid & 31;
  const int half = lane >> 4;
  const int lx   = lane & 15;
  const int mSub = wv & 3, nSub = wv >> 2;
  const int coRow0 = coB * 64 + mSub * 16;

  v8f acc[2][2];
  for (int row = 0; row < 2; ++row)
    for (int t = 0; t < 2; ++t)
      for (int r = 0; r < 8; ++r) {
        int co = coRow0 + half * 8 + r;
        acc[row][t][r] = b51[co] + b11[co];
      }

  const unsigned sA51base = lds_addr32(&sA51[0]);
  const unsigned sA11base = lds_addr32(&sA11[0]);

  for (int kk = 0; kk < 8; ++kk) {
    __syncthreads();
    // w51 taps: async, 1280 x b128
    for (int j = tid; j < 1280; j += 256) {
      int cip4 = j & 3;
      int rem = j >> 2;
      int co = rem / 5, kh = rem - co * 5;
      const bf16_t* src =
          w51b + ((size_t)((coB * 64 + co) * 5 + kh) * CC + kk * 32 + cip4 * 8);
      unsigned dst = sA51base + (unsigned)(((kh * 64 + co) * 32 + cip4 * 8) * 2);
      async_cp_b128(dst, src);
    }
    // w11: async, 256 x b128
    for (int j = tid; j < 256; j += 256) {
      int cip4 = j & 3;
      int co = j >> 2;
      const bf16_t* src =
          w11b + ((size_t)(coB * 64 + co) * CC + kk * 32 + cip4 * 8);
      unsigned dst = sA11base + (unsigned)((co * 32 + cip4 * 8) * 2);
      async_cp_b128(dst, src);
    }
    // t1 rows h0-2 .. h0+3 (6 rows), transposed, zero-pad in h
    for (int idx = tid; idx < 6 * 64 * 32; idx += 256) {
      int ci  = idx & 31;
      int rem = idx >> 5;                 // 0..383
      int n   = rem & 63;
      int row = rem >> 6;                 // 0..5
      int hh = h0 + row - 2;
      bf16_t v = (bf16_t)0.0f;
      if ((unsigned)hh < (unsigned)HH)
        v = t1b[(size_t)(b * CC + kk * 32 + ci) * HWW + hh * WW + w0 + n];
      sB51[(row * 64 + n) * SBS + ci] = v;
    }
    // x rows h0, h0+1 (1x1 branch), transposed
    for (int idx = tid; idx < 2 * 64 * 32; idx += 256) {
      int ci  = idx & 31;
      int rem = idx >> 5;
      int n   = rem & 63;
      int row = rem >> 6;                 // 0..1
      sB11[(row * 64 + n) * 36 + ci] =
          xb[(size_t)(b * CC + kk * 32 + ci) * HWW + (h0 + row) * WW + w0 + n];
    }
    wait_async0();
    __syncthreads();

    for (int kh = 0; kh < 5; ++kh) {
      FragBF a;
      const unsigned int* ap =
          (const unsigned int*)(sA51 + (kh * 64 + mSub * 16 + lx) * 32);
      #pragma unroll
      for (int i = 0; i < 8; ++i) {
        int kb = (i >> 2) * 16 + half * 8 + (i & 3) * 2;
        a.u[i] = ap[kb >> 1];
      }
      #pragma unroll
      for (int row = 0; row < 2; ++row) {
        #pragma unroll
        for (int t = 0; t < 2; ++t) {
          FragBF bb;
          int n = (nSub * 2 + t) * 16 + lx;
          const unsigned int* bp =
              (const unsigned int*)(sB51 + ((row + kh) * 64 + n) * SBS);
          #pragma unroll
          for (int i = 0; i < 8; ++i) {
            int kb = (i >> 2) * 16 + half * 8 + (i & 3) * 2;
            bb.u[i] = bp[kb >> 1];
          }
          acc[row][t] = __builtin_amdgcn_wmma_f32_16x16x32_bf16(
              false, a.v, false, bb.v, (short)0, acc[row][t], false, false);
        }
      }
    }
    {   // fused 1x1 tap
      FragBF a;
      const unsigned int* ap = (const unsigned int*)(sA11 + (mSub * 16 + lx) * 32);
      #pragma unroll
      for (int i = 0; i < 8; ++i) {
        int kb = (i >> 2) * 16 + half * 8 + (i & 3) * 2;
        a.u[i] = ap[kb >> 1];
      }
      #pragma unroll
      for (int row = 0; row < 2; ++row) {
        #pragma unroll
        for (int t = 0; t < 2; ++t) {
          FragBF bb;
          int n = (nSub * 2 + t) * 16 + lx;
          const unsigned int* bp =
              (const unsigned int*)(sB11 + (row * 64 + n) * 36);
          #pragma unroll
          for (int i = 0; i < 8; ++i) {
            int kb = (i >> 2) * 16 + half * 8 + (i & 3) * 2;
            bb.u[i] = bp[kb >> 1];
          }
          acc[row][t] = __builtin_amdgcn_wmma_f32_16x16x32_bf16(
              false, a.v, false, bb.v, (short)0, acc[row][t], false, false);
        }
      }
    }
  }

  #pragma unroll
  for (int row = 0; row < 2; ++row)
    #pragma unroll
    for (int t = 0; t < 2; ++t) {
      int w = w0 + (nSub * 2 + t) * 16 + lx;
      #pragma unroll
      for (int r = 0; r < 8; ++r) {
        int co = coRow0 + half * 8 + r;
        feat[(size_t)(b * CC + co) * HWW + (h0 + row) * WW + w] = acc[row][t][r];
      }
    }
}

// ------------- rotated feature align (points=1) + residual adds -------------
__global__ __launch_bounds__(256)
void k_align(const float* __restrict__ x, const float* __restrict__ feat,
             const float* __restrict__ boxes, float* __restrict__ out) {
  const int w0 = blockIdx.x * 64;
  const int h  = blockIdx.y;
  const int b  = blockIdx.z;
  const int tid = threadIdx.x;
  const int w  = w0 + (tid & 63);
  const int cg = tid >> 6;

  size_t pidx = (size_t)b * HWW + (size_t)h * WW + w;
  const float* bx = boxes + pidx * 5;
  float py = bx[0] * 0.125f;
  float px = bx[1] * 0.125f;
  bool valid = (py >= -1.0f) && (py <= (float)HH) && (px >= -1.0f) && (px <= (float)WW);
  py = fmaxf(py, 0.0f);
  px = fmaxf(px, 0.0f);
  int yl0 = (int)floorf(py), xl0 = (int)floorf(px);
  bool ey = yl0 >= HH - 1, ex = xl0 >= WW - 1;
  int yl = ey ? HH - 1 : yl0;
  int yh = ey ? HH - 1 : (yl0 + 1 < HH - 1 ? yl0 + 1 : HH - 1);
  int xl = ex ? WW - 1 : xl0;
  int xh = ex ? WW - 1 : (xl0 + 1 < WW - 1 ? xl0 + 1 : WW - 1);
  if (ey) py = (float)(HH - 1);
  if (ex) px = (float)(WW - 1);
  float ly = py - (float)yl, lxf = px - (float)xl;
  float hy = 1.0f - ly, hx = 1.0f - lxf;
  float wll = hy * hx, wlh = hy * lxf, whl = ly * hx, whh = ly * lxf;
  if (!valid) { wll = wlh = whl = whh = 0.0f; }

  for (int c = cg; c < CC; c += 4) {
    size_t cb = (size_t)(b * CC + c) * HWW;
    float v = wll * feat[cb + yl * WW + xl] + wlh * feat[cb + yl * WW + xh]
            + whl * feat[cb + yh * WW + xl] + whh * feat[cb + yh * WW + xh];
    size_t o = cb + (size_t)h * WW + w;
    out[o] = x[o] + feat[o] + v;
  }
}

// --------------------------------- launcher ---------------------------------
extern "C" void kernel_launch(void* const* d_in, const int* in_sizes, int n_in,
                              void* d_out, int out_size, void* d_ws, size_t ws_size,
                              hipStream_t stream) {
  // setup_inputs order: x, best_rbboxes, w51, b51, w15, b15, w11, b11
  const float* x     = (const float*)d_in[0];
  const float* boxes = (const float*)d_in[1];
  const float* w51   = (const float*)d_in[2];
  const float* b51   = (const float*)d_in[3];
  const float* w15   = (const float*)d_in[4];
  const float* b15   = (const float*)d_in[5];
  const float* w11   = (const float*)d_in[6];
  const float* b11   = (const float*)d_in[7];
  float* out = (float*)d_out;

  const size_t N = (size_t)2 * CC * HWW;      // 18,874,368 elems
  char* ws = (char*)d_ws;
  bf16_t* xb   = (bf16_t*)(ws);                     // 2N bytes
  bf16_t* t1b  = (bf16_t*)(ws + 2 * N);             // 2N bytes
  float*  feat = (float*) (ws + 4 * N);             // 4N bytes
  bf16_t* w15b = (bf16_t*)(ws + 8 * N);             // 655,360 B
  bf16_t* w51b = (bf16_t*)(ws + 8 * N + 655360);    // 655,360 B
  bf16_t* w11b = (bf16_t*)(ws + 8 * N + 1310720);   // 131,072 B

  k_cvt_x <<<(int)(N / 256), 256, 0, stream>>>(x, xb, (int)N);
  k_cvt_w5<<<1280, 256, 0, stream>>>(w15, w15b);
  k_cvt_w5<<<1280, 256, 0, stream>>>(w51, w51b);
  k_cvt_w1<<<256, 256, 0, stream>>>(w11, w11b);

  dim3 gA(3, 48, 8);    // w-tiles x h-blocks(4 rows) x (b*4 co-blocks)
  k_conv1x5<<<gA, 256, 0, stream>>>(xb, w15b, b15, t1b);

  dim3 gB(3, 96, 8);    // w-tiles x h-blocks(2 rows) x (b*4 co-blocks)
  k_conv5x1_1x1<<<gB, 256, 0, stream>>>(t1b, xb, w51b, w11b, b51, b11, feat);

  dim3 gAlign(3, 192, 2);
  k_align<<<gAlign, 256, 0, stream>>>(x, feat, boxes, out);
}
